// S6_8727373545524
// MI455X (gfx1250) — matrix operand order repaired and checked by reading, not compiled, not run on hardware
//
#include <hip/hip_runtime.h>
#include <hip/hip_bf16.h>

// ---------------------------------------------------------------------------
// S6 forward, algebraically collapsed:
//   y[t,d] = x[t,d] * softplus((x@W1+b1)[t,d]) * s[t]
//   s[t]   = sum_n (x@W2+b2)[t,n] * (x@W3+b3)[t,n]
// Heavy GEMM: v_wmma_f32_16x16x32_bf16 (wave32 CDNA5).
// Packed path: x and W1 pre-converted to bf16 once, laid out so every hot-loop
// operand fetch is a single b128 load (A: row-major K-pairs; B: K-major
// per-column pairs). Fallback path converts inline if workspace is small.
// ---------------------------------------------------------------------------

#define BS    4
#define LSEQ  2048
#define T_TOK (BS * LSEQ)   // 8192 tokens
#define DDIM  1024
#define KP    (DDIM / 2)    // packed K pairs per row/column
#define NDIM  16

typedef __attribute__((ext_vector_type(16))) __bf16 v16bf;
typedef __attribute__((ext_vector_type(2)))  __bf16 bf16x2;
typedef __attribute__((ext_vector_type(8)))  float  v8f;
typedef __attribute__((ext_vector_type(2)))  float  f32x2;

union ABf16 {
    unsigned int u[8];   // 8 VGPRs, 2 bf16 each
    uint4        q[2];
    v16bf        v;
};

// f32 pair -> packed bf16x2 (lowers to v_cvt_pk_bf16_f32).
__device__ __forceinline__ unsigned int pack_bf16(float lo, float hi) {
    bf16x2 p;
    p.x = (__bf16)lo;
    p.y = (__bf16)hi;
    return __builtin_bit_cast(unsigned int, p);
}

__device__ __forceinline__ float softplus_f(float z) {
    return (z > 20.0f) ? z : log1pf(__expf(z));
}

// Inline A tile (16 rows x 32 K of f32 x -> bf16) in contiguous K order (k = 16*hi + s).
__device__ __forceinline__ v16bf load_a_inline(const float* __restrict__ xrow,
                                               int k0, int ka) {
    ABf16 a;
#pragma unroll
    for (int j = 0; j < 4; ++j) {
        const f32x2 lo = *(const f32x2*)(xrow + k0 + ka + 2 * j);
        const f32x2 hi = *(const f32x2*)(xrow + k0 + ka + 2 * j + 8);
        a.u[j]     = pack_bf16(lo.x, lo.y);
        a.u[j + 4] = pack_bf16(hi.x, hi.y);
    }
    return a.v;
}

// Packed A tile: two contiguous b128 loads from Xp (row-major K pairs).
__device__ __forceinline__ v16bf load_a_packed(const unsigned int* __restrict__ xprow,
                                               int k0, int ka) {
    ABf16 a;
    a.q[0] = *(const uint4*)(xprow + (k0 >> 1) + (ka >> 1));
    a.q[1] = *(const uint4*)(xprow + (k0 >> 1) + 4 + (ka >> 1));
    return a.v;
}

// ---------------------------------------------------------------------------
// Pack kernels (one-shot, ~54 MB total traffic).
// ---------------------------------------------------------------------------

// Xp[t*KP + k2] = pack(x[t*D + 2k2], x[t*D + 2k2+1])  -- fully coalesced.
__global__ __launch_bounds__(256) void s6_pack_x(
    const float* __restrict__ x, unsigned int* __restrict__ Xp)
{
    const size_t tid = (size_t)blockIdx.x * blockDim.x + threadIdx.x; // < T*KP
    const f32x2 p = *(const f32x2*)(x + 2 * tid);
    Xp[tid] = pack_bf16(p.x, p.y);
}

// Wpt[n*KP + k2] = pack(W1[2k2*D + n], W1[(2k2+1)*D + n])
// Reads coalesced over n; scattered writes absorbed by L2 (2 MB total).
__global__ __launch_bounds__(256) void s6_pack_w1t(
    const float* __restrict__ W1, unsigned int* __restrict__ Wpt)
{
    const int tid = blockIdx.x * blockDim.x + threadIdx.x;  // < KP * D
    const int k2  = tid >> 10;          // packed-row index
    const int n   = tid & (DDIM - 1);   // column
    const size_t base = (size_t)(2 * k2) * DDIM + n;
    Wpt[(size_t)n * KP + k2] = pack_bf16(W1[base], W1[base + DDIM]);
}

// ---------------------------------------------------------------------------
// Kernel 1: per 16-token tile, B=X@W2 and C=X@W3 with WMMA, then
// s[t] = sum_n (B+b2)*(C+b3). One wave per tile; 512 waves.
// ---------------------------------------------------------------------------
template <bool PACKED>
__global__ __launch_bounds__(256) void s6_bc_dot(
    const float* __restrict__ x, const unsigned int* __restrict__ Xp,
    const float* __restrict__ W2, const float* __restrict__ b2,
    const float* __restrict__ W3, const float* __restrict__ b3,
    float* __restrict__ s_tok)
{
    const int lane = threadIdx.x & 31;
    const int wave = (blockIdx.x * blockDim.x + threadIdx.x) >> 5;   // 0..511
    const int row0 = wave * 16;

    const int m16 = lane & 15;
    const int hi  = lane >> 4;
    const int ka  = hi * 16;
    const int kb  = hi * 16;

    v8f accB = {};
    v8f accC = {};

    const float*        xrow  = x  + (size_t)(row0 + m16) * DDIM;
    const unsigned int* xprow = Xp + (size_t)(row0 + m16) * KP;

    for (int k0 = 0; k0 < DDIM; k0 += 32) {
        const v16bf a = PACKED ? load_a_packed(xprow, k0, ka)
                               : load_a_inline(xrow, k0, ka);
        ABf16 bB, bC;
#pragma unroll
        for (int j = 0; j < 8; ++j) {
            const int k = k0 + kb + 2 * j;
            bB.u[j] = pack_bf16(W2[(size_t)k * NDIM + m16],
                                W2[(size_t)(k + 1) * NDIM + m16]);
            bC.u[j] = pack_bf16(W3[(size_t)k * NDIM + m16],
                                W3[(size_t)(k + 1) * NDIM + m16]);
        }
        accB = __builtin_amdgcn_wmma_f32_16x16x32_bf16(
                   false, a, false, bB.v, (short)0, accB, false, false);
        accC = __builtin_amdgcn_wmma_f32_16x16x32_bf16(
                   false, a, false, bC.v, (short)0, accC, false, false);
    }

    const float bb = b2[m16];
    const float bc = b3[m16];
#pragma unroll
    for (int r = 0; r < 8; ++r) {
        // D layout: lanes 0-15 hold M=r, lanes 16-31 hold M=8+r, N=lane&15.
        float p = (accB[r] + bb) * (accC[r] + bc);
        p += __shfl_xor(p, 1, 32);
        p += __shfl_xor(p, 2, 32);
        p += __shfl_xor(p, 4, 32);
        p += __shfl_xor(p, 8, 32);
        if (m16 == 0) s_tok[row0 + hi * 8 + r] = p;
    }
}

// ---------------------------------------------------------------------------
// Kernel 2: Z = X @ W1 with WMMA; fused epilogue
//   y[t,d] = x[t,d] * softplus(Z[t,d] + b1[d]) * s[t]
// Each wave: one 16-row M tile x NT 16-col N tiles (A reused NT times).
// ---------------------------------------------------------------------------
template <bool PACKED, int NT>
__global__ __launch_bounds__(256) void s6_main(
    const float* __restrict__ x,
    const unsigned int* __restrict__ Xp,
    const float* __restrict__ W1,
    const unsigned int* __restrict__ Wpt,
    const float* __restrict__ b1,
    const float* __restrict__ s_tok,
    float* __restrict__ out)
{
    constexpr int GRP = 64 / NT;  // N-groups per M tile (64 col-tiles total)

    const int lane  = threadIdx.x & 31;
    const int wave  = (blockIdx.x * blockDim.x + threadIdx.x) >> 5;
    const int mtile = wave / GRP;
    const int ngrp  = wave % GRP;
    const int row0  = mtile * 16;
    const int col0  = ngrp * (16 * NT);

    const int m16 = lane & 15;
    const int hi  = lane >> 4;
    const int ka  = hi * 16;
    const int kb  = hi * 16;

    v8f acc[NT] = {};

    const float*        xrow  = x  + (size_t)(row0 + m16) * DDIM;
    const unsigned int* xprow = Xp + (size_t)(row0 + m16) * KP;

    for (int k0 = 0; k0 < DDIM; k0 += 32) {
        v16bf a;
        if (PACKED) {
            if (k0 + 32 < DDIM) __builtin_prefetch(xprow + ((k0 + 32) >> 1), 0, 3);
            a = load_a_packed(xprow, k0, ka);
        } else {
            if (k0 + 32 < DDIM) __builtin_prefetch(xrow + k0 + 32, 0, 3);
            a = load_a_inline(xrow, k0, ka);
        }

#pragma unroll
        for (int c = 0; c < NT; ++c) {
            ABf16 b;
            if (PACKED) {
                // B tile c: 8 contiguous K pairs of column col0+c*16+m16.
                const unsigned int* wcol =
                    Wpt + (size_t)(col0 + c * 16 + m16) * KP + (k0 >> 1) + hi * 8;
                b.q[0] = *(const uint4*)(wcol);
                b.q[1] = *(const uint4*)(wcol + 4);
            } else {
                const float* wbase =
                    W1 + (size_t)(k0 + kb) * DDIM + col0 + c * 16 + m16;
#pragma unroll
                for (int j = 0; j < 8; ++j)
                    b.u[j] = pack_bf16(wbase[(size_t)(2 * j) * DDIM],
                                       wbase[(size_t)(2 * j + 1) * DDIM]);
            }
            acc[c] = __builtin_amdgcn_wmma_f32_16x16x32_bf16(
                         false, a, false, b.v, (short)0, acc[c], false, false);
        }
    }

    // Fused epilogue.
#pragma unroll
    for (int c = 0; c < NT; ++c) {
        const int col    = col0 + c * 16 + m16;
        const float bcol = b1[col];
#pragma unroll
        for (int r = 0; r < 8; ++r) {
            const int t    = row0 + hi * 8 + r;
            const float z  = acc[c][r] + bcol;
            const float sp = softplus_f(z);
            out[(size_t)t * DDIM + col] = x[(size_t)t * DDIM + col] * sp * s_tok[t];
        }
    }
}

// ---------------------------------------------------------------------------
extern "C" void kernel_launch(void* const* d_in, const int* in_sizes, int n_in,
                              void* d_out, int out_size, void* d_ws, size_t ws_size,
                              hipStream_t stream)
{
    const float* x  = (const float*)d_in[0];
    const float* W1 = (const float*)d_in[1];
    const float* b1 = (const float*)d_in[2];
    const float* W2 = (const float*)d_in[3];
    const float* b2 = (const float*)d_in[4];
    const float* W3 = (const float*)d_in[5];
    const float* b3 = (const float*)d_in[6];
    // d_in[7] (A) is dead code in the reference (multiplied by zero state).

    float* out = (float*)d_out;

    // Workspace: [ s_tok 32 KB ][ Wpt 2 MB ][ Xp 16 MB ]
    const size_t S_BYTES  = (size_t)T_TOK * sizeof(float);
    const size_t WP_BYTES = (size_t)DDIM * KP * sizeof(unsigned int);
    const size_t XP_BYTES = (size_t)T_TOK * KP * sizeof(unsigned int);
    float*        s_tok = (float*)d_ws;
    unsigned int* Wpt   = (unsigned int*)((char*)d_ws + S_BYTES);
    unsigned int* Xp    = (unsigned int*)((char*)d_ws + S_BYTES + WP_BYTES);
    const bool packed   = (ws_size >= S_BYTES + WP_BYTES + XP_BYTES);

    if (packed) {
        s6_pack_x  <<<(size_t)T_TOK * KP / 256, 256, 0, stream>>>(x, Xp);
        s6_pack_w1t<<<(size_t)KP * DDIM / 256, 256, 0, stream>>>(W1, Wpt);

        s6_bc_dot<true><<<(T_TOK / 16) / 8, 256, 0, stream>>>(
            x, Xp, W2, b2, W3, b3, s_tok);

        // 512 M-tiles x 8 N-groups (NT=8) = 4096 waves, 8 waves/block.
        s6_main<true, 8><<<(T_TOK / 16) * 8 / 8, 256, 0, stream>>>(
            x, Xp, W1, Wpt, b1, s_tok, out);
    } else {
        s6_bc_dot<false><<<(T_TOK / 16) / 8, 256, 0, stream>>>(
            x, Xp, W2, b2, W3, b3, s_tok);

        // Fallback: inline conversion, NT=4 to limit register pressure.
        s6_main<false, 4><<<(T_TOK / 16) * 16 / 8, 256, 0, stream>>>(
            x, Xp, W1, Wpt, b1, s_tok, out);
    }
}
